// TransformerWithMoE_43748536877322
// MI455X (gfx1250) — matrix operand order repaired and checked by reading, not compiled, not run on hardware
//
#include <hip/hip_runtime.h>

// ---------------------------------------------------------------------------
// MI455X (gfx1250) implementation. wave32, WMMA bf16 16x16x32, f32 accumulate,
// async global->LDS staging (ASYNCcnt) for the GEMM weight/activation tiles.
// ---------------------------------------------------------------------------

typedef unsigned short u16;
typedef unsigned int   u32;

typedef __bf16 v16bf __attribute__((ext_vector_type(16)));
typedef float  v8f   __attribute__((ext_vector_type(8)));
typedef int    v4i   __attribute__((ext_vector_type(4)));

union BF16Frag { v16bf v; u32 u[8]; u16 h[16]; };

__device__ __forceinline__ u16 f32_bf16(float f) {
  union { float f; u32 u; } a; a.f = f;
  u32 r = a.u + 0x7FFFu + ((a.u >> 16) & 1u);   // round-to-nearest-even
  return (u16)(r >> 16);
}
__device__ __forceinline__ float bf16_f32(u16 h) {
  union { u32 u; float f; } a; a.u = ((u32)h) << 16;
  return a.f;
}

__device__ __forceinline__ v8f wmma_bf16(const BF16Frag& a, const BF16Frag& b, v8f c) {
  return __builtin_amdgcn_wmma_f32_16x16x32_bf16(false, a.v, false, b.v, (short)0, c, false, false);
}

// ---- async global -> LDS copy (16 bytes per lane), tracked by ASYNCcnt ----
typedef __attribute__((address_space(1))) v4i* gv4i_p;   // global (device) AS
typedef __attribute__((address_space(3))) v4i* lv4i_p;   // LDS AS

#if defined(__has_builtin) && __has_builtin(__builtin_amdgcn_global_load_async_to_lds_b128)
__device__ __forceinline__ void async_ld128(const void* g, void* l) {
  __builtin_amdgcn_global_load_async_to_lds_b128((gv4i_p)g, (lv4i_p)l, 0, 0);
}
#else
__device__ __forceinline__ void async_ld128(const void* g, void* l) {
  __attribute__((address_space(3))) char* lp = (__attribute__((address_space(3))) char*)l;
  unsigned loff = (unsigned)(unsigned long long)lp;
  asm volatile("global_load_async_to_lds_b128 %0, %1, off"
               :: "v"(loff), "v"(g) : "memory");
}
#endif

__device__ __forceinline__ void async_wait0() {
#if defined(__has_builtin) && __has_builtin(__builtin_amdgcn_s_wait_asynccnt)
  __builtin_amdgcn_s_wait_asynccnt(0);
#else
  asm volatile("s_wait_asynccnt 0" ::: "memory");
#endif
}

// A-fragment (16x32 bf16) from LDS row base. Lane (l,hi) holds row=l,
// K in {16*hi .. 16*hi+15}.
__device__ __forceinline__ void load_a_lds(BF16Frag& f, const u16* rowbase, int hi) {
  const u32* p0 = (const u32*)(rowbase + 16 * hi);
  const u32* p1 = (const u32*)(rowbase + 16 * hi + 8);
#pragma unroll
  for (int i = 0; i < 4; i++) { f.u[i] = p0[i]; f.u[4 + i] = p1[i]; }
}
// B-fragment (32x16 bf16) from LDS column base (k contiguous). Lane holds
// col=l, K in {16*hi .. 16*hi+15}.
__device__ __forceinline__ void load_b_lds(BF16Frag& f, const u16* colbase, int hi) {
  const u32* p = (const u32*)(colbase + 16 * hi);
#pragma unroll
  for (int i = 0; i < 8; i++) f.u[i] = p[i];
}
// A-fragment directly from global row-major bf16 (16B aligned chunks).
__device__ __forceinline__ void load_a_g(BF16Frag& f, const u16* rowbase, int hi) {
  uint4 a = *(const uint4*)(rowbase + 16 * hi);
  uint4 b = *(const uint4*)(rowbase + 16 * hi + 8);
  f.u[0] = a.x; f.u[1] = a.y; f.u[2] = a.z; f.u[3] = a.w;
  f.u[4] = b.x; f.u[5] = b.y; f.u[6] = b.z; f.u[7] = b.w;
}
// B-fragment from global where the 16 K values are contiguous (32B).
__device__ __forceinline__ void load16_g(BF16Frag& f, const u16* p) {
  uint4 a = *(const uint4*)p;
  uint4 b = *(const uint4*)(p + 8);
  f.u[0] = a.x; f.u[1] = a.y; f.u[2] = a.z; f.u[3] = a.w;
  f.u[4] = b.x; f.u[5] = b.y; f.u[6] = b.z; f.u[7] = b.w;
}

// ---------------------------------------------------------------------------
// Kernel 0: weight pack: W f32 [K,N] row-major -> BT bf16 [N,K] row-major.
// One-time ~216MB of traffic (~9us at 23.3 TB/s) buys conversion-free,
// fully-async staging in every GEMM hot loop.
// ---------------------------------------------------------------------------
__global__ __launch_bounds__(256) void pack_bt_k(
    const float* __restrict__ W, u16* __restrict__ BT, int K, int N) {
  __shared__ u16 tile[64][72];
  const int tid = threadIdx.x;
  const int n0 = blockIdx.x * 64, k0 = blockIdx.y * 64;
  const size_t eoff = (size_t)blockIdx.z * K * N;
  const float* Wp = W + eoff;
  u16* Bp = BT + eoff;
#pragma unroll
  for (int it = 0; it < 4; it++) {
    int idx = it * 256 + tid;        // 0..1023
    int kk = idx >> 4;               // 0..63
    int n4 = (idx & 15) * 4;         // 0..60
    float4 f = *(const float4*)(Wp + (size_t)(k0 + kk) * N + n0 + n4);
    tile[n4 + 0][kk] = f32_bf16(f.x);
    tile[n4 + 1][kk] = f32_bf16(f.y);
    tile[n4 + 2][kk] = f32_bf16(f.z);
    tile[n4 + 3][kk] = f32_bf16(f.w);
  }
  __syncthreads();
#pragma unroll
  for (int it = 0; it < 8; it++) {
    int idx = it * 256 + tid;        // 0..2047
    int n = idx >> 5;                // 0..63
    int kp = idx & 31;               // k pair
    u32 w = (u32)tile[n][2 * kp] | ((u32)tile[n][2 * kp + 1] << 16);
    *(u32*)(Bp + (size_t)(n0 + n) * K + k0 + 2 * kp) = w;
  }
}

// ---------------------------------------------------------------------------
// Kernel 1: embedding gather + LayerNorm1 -> y1 (bf16), emb (f32 residual)
// ---------------------------------------------------------------------------
__global__ __launch_bounds__(256) void embed_ln_k(
    const int* __restrict__ x, const float* __restrict__ tab,
    const float* __restrict__ g, const float* __restrict__ be,
    float* __restrict__ emb, u16* __restrict__ y) {
  __shared__ float rs_[256], rq_[256];
  const int t = blockIdx.x, tid = threadIdx.x;
  const float* row = tab + (size_t)x[t] * 1024;
  float4 v = *((const float4*)row + tid);
  rs_[tid] = v.x + v.y + v.z + v.w;
  rq_[tid] = v.x * v.x + v.y * v.y + v.z * v.z + v.w * v.w;
  __syncthreads();
  for (int st = 128; st > 0; st >>= 1) {
    if (tid < st) { rs_[tid] += rs_[tid + st]; rq_[tid] += rq_[tid + st]; }
    __syncthreads();
  }
  const float mean = rs_[0] * (1.f / 1024.f);
  const float var  = rq_[0] * (1.f / 1024.f) - mean * mean;
  const float rstd = rsqrtf(var + 1e-5f);
  *((float4*)(emb + (size_t)t * 1024) + tid) = v;
  const int d0 = tid * 4;
  float vv[4] = {v.x, v.y, v.z, v.w};
#pragma unroll
  for (int c = 0; c < 4; c++) {
    int d = d0 + c;
    y[(size_t)t * 1024 + d] = f32_bf16((vv[c] - mean) * rstd * g[d] + be[d]);
  }
}

// ---------------------------------------------------------------------------
// Kernel 2: bf16 WMMA GEMM: C[M,N] = A_bf16[M,K] @ BT_bf16[N,K]^T + bias
//           (+ optional f32 residual). 128 thr = 4 waves, 128x128 tile,
//           wave owns 64x64 (16 wmma / 32-k-step). Tiles staged with
//           per-lane async b128 copies (ASYNCcnt).
// ---------------------------------------------------------------------------
template <bool OUT_BF16, bool RES>
__global__ __launch_bounds__(128) void gemm_bf16_k(
    const u16* __restrict__ A, const u16* __restrict__ BT,
    const float* __restrict__ bias, const float* __restrict__ Res,
    u16* __restrict__ Cb, float* __restrict__ Cf, int M, int N, int K) {
  __shared__ u16 Al[128][40];   // 80B row stride: 16B-aligned, 20-bank skew
  __shared__ u16 Bl[128][40];
  const int tid = threadIdx.x;
  const int lane = tid & 31, wv = tid >> 5;
  const int l = lane & 15, hi = lane >> 4;
  const int wm = wv >> 1, wn = wv & 1;
  const int n0 = blockIdx.x * 128, m0 = blockIdx.y * 128;

  const v8f vz = {0.f, 0.f, 0.f, 0.f, 0.f, 0.f, 0.f, 0.f};
  v8f acc[4][4];
#pragma unroll
  for (int i = 0; i < 4; i++)
#pragma unroll
    for (int j = 0; j < 4; j++) acc[i][j] = vz;

  const u16* arow = A + (size_t)(m0 + tid) * K;
  const u16* brow = BT + (size_t)(n0 + tid) * K;

  for (int kb = 0; kb < K; kb += 32) {
    __syncthreads();   // previous tile fully consumed
#pragma unroll
    for (int q = 0; q < 4; q++) {
      async_ld128(arow + kb + 8 * q, &Al[tid][8 * q]);
      async_ld128(brow + kb + 8 * q, &Bl[tid][8 * q]);
    }
    __builtin_prefetch(brow + kb + 32, 0, 1);
    async_wait0();
    __syncthreads();
    BF16Frag af[4], bf_[4];
#pragma unroll
    for (int i = 0; i < 4; i++) load_a_lds(af[i], &Al[wm * 64 + 16 * i + l][0], hi);
#pragma unroll
    for (int j = 0; j < 4; j++) load_b_lds(bf_[j], &Bl[wn * 64 + 16 * j + l][0], hi);
#pragma unroll
    for (int i = 0; i < 4; i++)
#pragma unroll
      for (int j = 0; j < 4; j++) acc[i][j] = wmma_bf16(af[i], bf_[j], acc[i][j]);
  }

#pragma unroll
  for (int j = 0; j < 4; j++) {
    const int n = n0 + wn * 64 + 16 * j + l;
    const float bvv = bias ? bias[n] : 0.f;
#pragma unroll
    for (int i = 0; i < 4; i++) {
#pragma unroll
      for (int r = 0; r < 8; r++) {
        const int m = m0 + wm * 64 + 16 * i + 8 * hi + r;
        float v = acc[i][j][r] + bvv;
        if (RES) v += Res[(size_t)m * N + n];
        if (OUT_BF16) Cb[(size_t)m * N + n] = f32_bf16(v);
        else          Cf[(size_t)m * N + n] = v;
      }
    }
  }
}

// ---------------------------------------------------------------------------
// Kernel 3: fused attention, one block per (b, h, 64 q-rows), 4 waves.
// Two-pass softmax (pass1: row max; pass2: exp + P@V), all GEMMs via WMMA.
// ---------------------------------------------------------------------------
__global__ __launch_bounds__(128) void attn_k(
    const u16* __restrict__ Q, const u16* __restrict__ Kk,
    const u16* __restrict__ V, u16* __restrict__ CTX) {
  __shared__ u16 pbuf[4][16][34];   // per-wave P tile (16 rows x 32 keys)
  __shared__ u16 vt[64][34];        // V tile transposed: vt[dh][key]
  const int tid = threadIdx.x, lane = tid & 31, wv = tid >> 5;
  const int l = lane & 15, hi = lane >> 4;
  const int qt = blockIdx.x, h = blockIdx.y, b = blockIdx.z;
  const float sc = 0.125f;  // 1/sqrt(64)
  const v8f vz = {0.f, 0.f, 0.f, 0.f, 0.f, 0.f, 0.f, 0.f};

  const int trow = qt * 64 + wv * 16 + l;   // this lane's q row (A-matrix row)
  const u16* qrow = Q + (size_t)(b * 1024 + trow) * 1024 + h * 64;
  BF16Frag aq[2];
  load_a_g(aq[0], qrow, hi);
  load_a_g(aq[1], qrow + 32, hi);

  // ---- pass 1: row max of raw scores ----
  float rmax[8];
#pragma unroll
  for (int r = 0; r < 8; r++) rmax[r] = -3.0e38f;
  for (int kt = 0; kt < 1024; kt += 32) {
#pragma unroll
    for (int nt = 0; nt < 2; nt++) {
      const int key = kt + nt * 16 + l;
      const u16* krow = Kk + (size_t)(b * 1024 + key) * 1024 + h * 64;
      BF16Frag bk0, bk1;
      load16_g(bk0, krow + 16 * hi);
      load16_g(bk1, krow + 32 + 16 * hi);
      v8f s = vz;
      s = wmma_bf16(aq[0], bk0, s);
      s = wmma_bf16(aq[1], bk1, s);
#pragma unroll
      for (int r = 0; r < 8; r++) rmax[r] = fmaxf(rmax[r], s[r]);
    }
  }
#pragma unroll
  for (int r = 0; r < 8; r++) {
    float m = rmax[r];
    m = fmaxf(m, __shfl_xor(m, 1));
    m = fmaxf(m, __shfl_xor(m, 2));
    m = fmaxf(m, __shfl_xor(m, 4));
    m = fmaxf(m, __shfl_xor(m, 8));
    rmax[r] = m * sc;
  }

  // ---- pass 2: p = exp(s*sc - max); ctx += p @ V ----
  float rsum[8];
#pragma unroll
  for (int r = 0; r < 8; r++) rsum[r] = 0.f;
  v8f ctx[4];
#pragma unroll
  for (int j = 0; j < 4; j++) ctx[j] = vz;

  for (int kt = 0; kt < 1024; kt += 32) {
    __syncthreads();   // protect vt/pbuf from previous iteration's readers
    // stage V tile transposed: vt[dh][key], 32 keys x 64 dh
    for (int idx = tid; idx < 1024; idx += 128) {
      const int k = idx >> 5;            // key 0..31
      const int d2 = (idx & 31) * 2;     // dh pair
      u32 w = *(const u32*)(V + (size_t)(b * 1024 + kt + k) * 1024 + h * 64 + d2);
      vt[d2][k] = (u16)w;
      vt[d2 + 1][k] = (u16)(w >> 16);
    }
    // scores -> exp -> pbuf (C-layout row M = 8*hi + r, col N = lane)
#pragma unroll
    for (int nt = 0; nt < 2; nt++) {
      const int key = kt + nt * 16 + l;
      const u16* krow = Kk + (size_t)(b * 1024 + key) * 1024 + h * 64;
      BF16Frag bk0, bk1;
      load16_g(bk0, krow + 16 * hi);
      load16_g(bk1, krow + 32 + 16 * hi);
      v8f s = vz;
      s = wmma_bf16(aq[0], bk0, s);
      s = wmma_bf16(aq[1], bk1, s);
#pragma unroll
      for (int r = 0; r < 8; r++) {
        float p = __expf(s[r] * sc - rmax[r]);
        rsum[r] += p;
        pbuf[wv][8 * hi + r][nt * 16 + l] = f32_bf16(p);
      }
    }
    __syncthreads();
    BF16Frag ap;
    load_a_lds(ap, &pbuf[wv][l][0], hi);
#pragma unroll
    for (int j = 0; j < 4; j++) {
      BF16Frag bv_;
      load_b_lds(bv_, &vt[16 * j + l][0], hi);
      ctx[j] = wmma_bf16(ap, bv_, ctx[j]);
    }
  }

  float inv[8];
#pragma unroll
  for (int r = 0; r < 8; r++) {
    float sm = rsum[r];
    sm += __shfl_xor(sm, 1);
    sm += __shfl_xor(sm, 2);
    sm += __shfl_xor(sm, 4);
    sm += __shfl_xor(sm, 8);
    inv[r] = 1.f / sm;
  }
#pragma unroll
  for (int j = 0; j < 4; j++)
#pragma unroll
    for (int r = 0; r < 8; r++) {
      const int tok = qt * 64 + wv * 16 + 8 * hi + r;
      const int col = h * 64 + 16 * j + l;
      CTX[(size_t)(b * 1024 + tok) * 1024 + col] = f32_bf16(ctx[j][r] * inv[r]);
    }
}

// ---------------------------------------------------------------------------
// Kernel 4: LayerNorm2 + router softmax + top-2 -> y2 (bf16), gates[T,8]
// ---------------------------------------------------------------------------
__global__ __launch_bounds__(256) void ln2_router_k(
    const float* __restrict__ yin, const float* __restrict__ g,
    const float* __restrict__ be, const float* __restrict__ wr,
    const float* __restrict__ br, u16* __restrict__ y2,
    float* __restrict__ gates) {
  __shared__ float red[256][8];
  const int t = blockIdx.x, tid = threadIdx.x;
  float4 v = *((const float4*)(yin + (size_t)t * 1024) + tid);
  red[tid][0] = v.x + v.y + v.z + v.w;
  red[tid][1] = v.x * v.x + v.y * v.y + v.z * v.z + v.w * v.w;
  __syncthreads();
  for (int st = 128; st > 0; st >>= 1) {
    if (tid < st) { red[tid][0] += red[tid + st][0]; red[tid][1] += red[tid + st][1]; }
    __syncthreads();
  }
  const float mean = red[0][0] * (1.f / 1024.f);
  const float var  = red[0][1] * (1.f / 1024.f) - mean * mean;
  const float rstd = rsqrtf(var + 1e-5f);
  __syncthreads();   // red about to be reused for router partials

  const int d0 = tid * 4;
  float vv[4] = {v.x, v.y, v.z, v.w};
  float yn[4];
  float p[8];
#pragma unroll
  for (int e = 0; e < 8; e++) p[e] = 0.f;
#pragma unroll
  for (int c = 0; c < 4; c++) {
    const int d = d0 + c;
    yn[c] = (vv[c] - mean) * rstd * g[d] + be[d];
    y2[(size_t)t * 1024 + d] = f32_bf16(yn[c]);
#pragma unroll
    for (int e = 0; e < 8; e++) p[e] += yn[c] * wr[(size_t)d * 8 + e];
  }
#pragma unroll
  for (int e = 0; e < 8; e++) red[tid][e] = p[e];
  __syncthreads();
  for (int st = 128; st > 0; st >>= 1) {
    if (tid < st)
#pragma unroll
      for (int e = 0; e < 8; e++) red[tid][e] += red[tid + st][e];
    __syncthreads();
  }
  if (tid == 0) {
    float lg[8], mx = -3.0e38f;
#pragma unroll
    for (int e = 0; e < 8; e++) { lg[e] = red[0][e] + br[e]; mx = fmaxf(mx, lg[e]); }
    float ex[8], sum = 0.f;
#pragma unroll
    for (int e = 0; e < 8; e++) { ex[e] = __expf(lg[e] - mx); sum += ex[e]; }
    const float invs = 1.f / sum;
    float pr[8];
#pragma unroll
    for (int e = 0; e < 8; e++) pr[e] = ex[e] * invs;
    int i0 = 0;
    for (int e = 1; e < 8; e++) if (pr[e] > pr[i0]) i0 = e;
    int i1 = (i0 == 0) ? 1 : 0;
    for (int e = 0; e < 8; e++) if (e != i0 && pr[e] > pr[i1]) i1 = e;
#pragma unroll
    for (int e = 0; e < 8; e++)
      gates[(size_t)t * 8 + e] = (e == i0 || e == i1) ? pr[e] : 0.f;
  }
}

// ---------------------------------------------------------------------------
// Kernel 5: MoE. out[m,n] = sum_e gate[m,e] * (y2[m,:] @ We[:,n] + be[e,n]).
// Gate folded into A during staging (f32 multiply, re-rounded to bf16), single
// accumulator over all 8 experts, no atomics, single deterministic store.
// Weight stream (dominant, 64MB bf16) staged via async b128 copies.
// ---------------------------------------------------------------------------
__global__ __launch_bounds__(128) void moe_k(
    const u16* __restrict__ Y2, const u16* __restrict__ BTex,
    const float* __restrict__ Bex, const float* __restrict__ gates,
    float* __restrict__ Out) {
  __shared__ u16 Al[128][40];
  __shared__ u16 Bl[128][40];
  __shared__ float gl[128][8];
  const int tid = threadIdx.x;
  const int lane = tid & 31, wv = tid >> 5;
  const int l = lane & 15, hi = lane >> 4;
  const int wm = wv >> 1, wn = wv & 1;
  const int n0 = blockIdx.x * 128, m0 = blockIdx.y * 128;

  for (int idx = tid; idx < 1024; idx += 128)
    gl[idx >> 3][idx & 7] = gates[(size_t)(m0 + (idx >> 3)) * 8 + (idx & 7)];

  const v8f vz = {0.f, 0.f, 0.f, 0.f, 0.f, 0.f, 0.f, 0.f};
  v8f acc[4][4];
#pragma unroll
  for (int i = 0; i < 4; i++)
#pragma unroll
    for (int j = 0; j < 4; j++) acc[i][j] = vz;

  const u16* arow = Y2 + (size_t)(m0 + tid) * 1024;

  for (int e = 0; e < 8; e++) {
    const u16* brow = BTex + ((size_t)e * 4096 + n0 + tid) * 1024;
    for (int kb = 0; kb < 1024; kb += 32) {
      __syncthreads();   // previous tile consumed (also orders gl stage once)
      // async-stage expert weight tile rows (64B per lane)
#pragma unroll
      for (int q = 0; q < 4; q++)
        async_ld128(brow + kb + 8 * q, &Bl[tid][8 * q]);
      __builtin_prefetch(brow + kb + 32, 0, 1);
      // stage A scaled by this row's gate for expert e (overlaps async)
      {
        const float gr = gl[tid][e];
        const u32* src = (const u32*)(arow + kb);
        u32* dst = (u32*)&Al[tid][0];
#pragma unroll
        for (int q = 0; q < 16; q++) {
          const u32 w = src[q];
          const float lo = bf16_f32((u16)w) * gr;
          const float hv = bf16_f32((u16)(w >> 16)) * gr;
          dst[q] = (u32)f32_bf16(lo) | ((u32)f32_bf16(hv) << 16);
        }
      }
      async_wait0();
      __syncthreads();
      BF16Frag af[4], bf_[4];
#pragma unroll
      for (int i = 0; i < 4; i++) load_a_lds(af[i], &Al[wm * 64 + 16 * i + l][0], hi);
#pragma unroll
      for (int j = 0; j < 4; j++) load_b_lds(bf_[j], &Bl[wn * 64 + 16 * j + l][0], hi);
#pragma unroll
      for (int i = 0; i < 4; i++)
#pragma unroll
        for (int j = 0; j < 4; j++) acc[i][j] = wmma_bf16(af[i], bf_[j], acc[i][j]);
    }
  }

  // epilogue: add sum_e gate[m,e]*be[e,n], single store
#pragma unroll
  for (int j = 0; j < 4; j++) {
    const int n = n0 + wn * 64 + 16 * j + l;
    float bev[8];
#pragma unroll
    for (int e = 0; e < 8; e++) bev[e] = Bex[(size_t)e * 4096 + n];
#pragma unroll
    for (int i = 0; i < 4; i++) {
#pragma unroll
      for (int r = 0; r < 8; r++) {
        const int row = wm * 64 + 16 * i + 8 * hi + r;
        float vsum = acc[i][j][r];
#pragma unroll
        for (int e = 0; e < 8; e++) vsum += gl[row][e] * bev[e];
        Out[(size_t)(m0 + row) * 4096 + n] = vsum;
      }
    }
  }
}

// ---------------------------------------------------------------------------
extern "C" void kernel_launch(void* const* d_in, const int* in_sizes, int n_in,
                              void* d_out, int out_size, void* d_ws, size_t ws_size,
                              hipStream_t stream) {
  (void)in_sizes; (void)n_in; (void)out_size; (void)ws_size;
  const int*   x   = (const int*)d_in[0];
  const float* tab = (const float*)d_in[1];
  const float* wq  = (const float*)d_in[2];
  const float* bq  = (const float*)d_in[3];
  const float* wk  = (const float*)d_in[4];
  const float* bk  = (const float*)d_in[5];
  const float* wvw = (const float*)d_in[6];
  const float* bv  = (const float*)d_in[7];
  const float* wpr = (const float*)d_in[8];
  const float* bpr = (const float*)d_in[9];
  const float* g1  = (const float*)d_in[10];
  const float* be1 = (const float*)d_in[11];
  const float* g2  = (const float*)d_in[12];
  const float* be2 = (const float*)d_in[13];
  const float* wr  = (const float*)d_in[14];
  const float* br  = (const float*)d_in[15];
  const float* wex = (const float*)d_in[16];
  const float* bex = (const float*)d_in[17];

  char* ws = (char*)d_ws;
  const size_t MB = 1024 * 1024;
  float* emb   = (float*)(ws + 0 * MB);    // 16 MB f32 [4096,1024]
  u16*   y1    = (u16*)  (ws + 16 * MB);   //  8 MB bf16
  u16*   qb    = (u16*)  (ws + 24 * MB);   //  8 MB
  u16*   kb_   = (u16*)  (ws + 32 * MB);   //  8 MB
  u16*   vb    = (u16*)  (ws + 40 * MB);   //  8 MB
  u16*   ctx   = (u16*)  (ws + 48 * MB);   //  8 MB
  float* proj  = (float*)(ws + 56 * MB);   // 16 MB f32
  u16*   y2    = (u16*)  (ws + 72 * MB);   //  8 MB
  float* gates = (float*)(ws + 80 * MB);   // 128 KB (1 MB reserved)
  u16*   btq   = (u16*)  (ws + 81 * MB);   //  2 MB bf16 [1024,1024]
  u16*   btk   = (u16*)  (ws + 83 * MB);   //  2 MB
  u16*   btv   = (u16*)  (ws + 85 * MB);   //  2 MB
  u16*   btp   = (u16*)  (ws + 87 * MB);   //  2 MB
  u16*   btex  = (u16*)  (ws + 89 * MB);   // 64 MB bf16 [8][4096][1024]

  // weight packing (independent of activations)
  dim3 pk1(16, 16);
  pack_bt_k<<<pk1, 256, 0, stream>>>(wq,  btq, 1024, 1024);
  pack_bt_k<<<pk1, 256, 0, stream>>>(wk,  btk, 1024, 1024);
  pack_bt_k<<<pk1, 256, 0, stream>>>(wvw, btv, 1024, 1024);
  pack_bt_k<<<pk1, 256, 0, stream>>>(wpr, btp, 1024, 1024);
  pack_bt_k<<<dim3(64, 16, 8), 256, 0, stream>>>(wex, btex, 1024, 4096);

  embed_ln_k<<<4096, 256, 0, stream>>>(x, tab, g1, be1, emb, y1);

  dim3 gqkv(1024 / 128, 4096 / 128);
  gemm_bf16_k<true, false><<<gqkv, 128, 0, stream>>>(y1, btq, bq, nullptr, qb, nullptr, 4096, 1024, 1024);
  gemm_bf16_k<true, false><<<gqkv, 128, 0, stream>>>(y1, btk, bk, nullptr, kb_, nullptr, 4096, 1024, 1024);
  gemm_bf16_k<true, false><<<gqkv, 128, 0, stream>>>(y1, btv, bv, nullptr, vb, nullptr, 4096, 1024, 1024);

  attn_k<<<dim3(16, 16, 4), 128, 0, stream>>>(qb, kb_, vb, ctx);

  gemm_bf16_k<false, true><<<gqkv, 128, 0, stream>>>(ctx, btp, bpr, emb, nullptr, proj, 4096, 1024, 1024);

  ln2_router_k<<<4096, 256, 0, stream>>>(proj, g2, be2, wr, br, y2, gates);

  moe_k<<<dim3(4096 / 128, 4096 / 128), 128, 0, stream>>>(y2, btex, bex, gates, (float*)d_out);
}
